// GATPredictor_69423851373033
// MI455X (gfx1250) — matrix operand order repaired and checked by reading, not compiled, not run on hardware
//
#include <hip/hip_runtime.h>
#include <math.h>

// ---------------- problem constants ----------------
#define NNODE   20000
#define FIN     128
#define HID     128
#define HEADS   8
#define F1      1024      // HEADS*HID
#define NCLS    18
#define NPAD    32        // padded layer-2 width
#define ERAW    200000
#define ETOT    220000    // + self loops
#define SLOPE   0.2f

typedef __attribute__((ext_vector_type(16))) __bf16 v16bf;
typedef __attribute__((ext_vector_type(8)))  __bf16 v8bf;
typedef __attribute__((ext_vector_type(8)))  float  v8f;

__device__ __forceinline__ __bf16 f2bf(float f) {
    union { float f; unsigned u; } in; in.f = f;
    unsigned r = (in.u + 0x7FFFu + ((in.u >> 16) & 1u)) >> 16;
    union { unsigned short s; __bf16 b; } out; out.s = (unsigned short)r;
    return out.b;
}

// max on float via signed/unsigned atomic trick (valid with -inf init)
__device__ __forceinline__ void atomicMaxF32(float* addr, float val) {
    if (val >= 0.0f) atomicMax((int*)addr, __float_as_int(val));
    else             atomicMin((unsigned int*)addr, __float_as_uint(val));
}

__device__ __forceinline__ void edge_pair(const int* ei, int e, int& src, int& dst) {
    if (e < ERAW) { src = ei[e]; dst = ei[ERAW + e]; }
    else          { src = dst = e - ERAW; }
}

// ---------------- conversion / packing ----------------
__global__ void k_cvt_bf16(const float* __restrict__ in, __bf16* __restrict__ out, int n) {
    int i = blockIdx.x * 256 + threadIdx.x;
    if (i < n) out[i] = f2bf(in[i]);
}

// pack W (row-major [K][ncols]) into B-fragment layout:
// chunk c = (ks*nTileCols + n)*2 + hi  -> 16 bf16 = W[ks*32+hi*16+j][n]
__global__ void k_pack_w(const float* __restrict__ W, __bf16* __restrict__ Wp,
                         int ksteps, int ncolsPad, int ncolsReal, int ldw) {
    int c = blockIdx.x * 256 + threadIdx.x;
    if (c >= ksteps * ncolsPad * 2) return;
    int hi = c & 1;
    int n  = (c >> 1) % ncolsPad;
    int ks = (c >> 1) / ncolsPad;
    __bf16* dst = Wp + (size_t)c * 16;
#pragma unroll
    for (int j = 0; j < 16; ++j) {
        int k = ks * 32 + hi * 16 + j;
        float v = (n < ncolsReal) ? W[(size_t)k * ldw + n] : 0.0f;
        dst[j] = f2bf(v);
    }
}

// ---------------- GEMM 1: H1 = Xb @ W1p  (M=20000,K=128,N=1024) ----------------
__global__ void __launch_bounds__(256)
k_gemm1(const __bf16* __restrict__ Xb, const __bf16* __restrict__ W1p,
        float* __restrict__ H1) {
    __shared__ __align__(16) __bf16 Alds[16 * FIN];
    const int tileM   = blockIdx.x * 16;
    const int colBase = blockIdx.y * 128;
    const int t = threadIdx.x;
    // stage 16x128 bf16 A tile (4 KB) once for all 8 waves
    *(v8bf*)(Alds + t * 8) = *(const v8bf*)(Xb + (size_t)tileM * FIN + t * 8);
    __syncthreads();

    const int w = t >> 5, lane = t & 31;
    const int l = lane & 15, hi = lane >> 4;
    const int n = colBase + w * 16 + l;

    v8f acc = {};
#pragma unroll
    for (int ks = 0; ks < 4; ++ks) {
        const int k0 = ks * 32;
        v8bf alo = *(const v8bf*)(Alds + l * FIN + k0 + 16 * hi);
        v8bf ahi = *(const v8bf*)(Alds + l * FIN + k0 + 16 * hi + 8);
        v16bf a;
#pragma unroll
        for (int i = 0; i < 8; ++i) { a[i] = alo[i]; a[i + 8] = ahi[i]; }
        v16bf b = *(const v16bf*)(W1p + ((((size_t)ks * F1 + n) * 2) + hi) * 16);
        acc = __builtin_amdgcn_wmma_f32_16x16x32_bf16(false, a, false, b,
                                                      (short)0, acc, false, false);
    }
    float* out = H1 + (size_t)tileM * F1 + colBase + w * 16 + l;
#pragma unroll
    for (int r = 0; r < 8; ++r) out[(size_t)(r + 8 * hi) * F1] = acc[r];
}

// ---------------- GEMM 2: H2 = X2b @ W2p  (M=20000,K=1024,N=32pad) ----------------
__global__ void __launch_bounds__(256)
k_gemm2(const __bf16* __restrict__ X2b, const __bf16* __restrict__ W2p,
        float* __restrict__ H2) {
    const int t = threadIdx.x;
    const int w = t >> 5, lane = t & 31;
    const int l = lane & 15, hi = lane >> 4;
    const int mTile = blockIdx.x * 4 + (w >> 1);   // wave-uniform guard
    const int nT    = w & 1;
    if (mTile >= NNODE / 16) return;
    const int m = mTile * 16 + l;
    const int n = nT * 16 + l;
    const __bf16* arow = X2b + (size_t)m * F1;

    v8f acc = {};
    for (int ks = 0; ks < 32; ++ks) {
        const int k0 = ks * 32;
        __builtin_prefetch(arow + k0 + 128, 0, 0);
        v8bf alo = *(const v8bf*)(arow + k0 + 16 * hi);
        v8bf ahi = *(const v8bf*)(arow + k0 + 16 * hi + 8);
        v16bf a;
#pragma unroll
        for (int i = 0; i < 8; ++i) { a[i] = alo[i]; a[i + 8] = ahi[i]; }
        v16bf b = *(const v16bf*)(W2p + ((((size_t)ks * NPAD + n) * 2) + hi) * 16);
        acc = __builtin_amdgcn_wmma_f32_16x16x32_bf16(false, a, false, b,
                                                      (short)0, acc, false, false);
    }
    float* out = H2 + (size_t)mTile * 16 * NPAD + nT * 16 + l;
#pragma unroll
    for (int r = 0; r < 8; ++r) out[(size_t)(r + 8 * hi) * NPAD] = acc[r];
}

// ---------------- attention scalars ----------------
__global__ void k_alpha1(const float* __restrict__ H1, const float* __restrict__ a_s,
                         const float* __restrict__ a_d,
                         float* __restrict__ as, float* __restrict__ ad) {
    int i = blockIdx.x * 256 + threadIdx.x;     // n*8 + h
    if (i >= NNODE * HEADS) return;
    int n = i >> 3, h = i & 7;
    const float* hp = H1 + (size_t)n * F1 + h * HID;
    const float* s = a_s + h * HID;
    const float* d = a_d + h * HID;
    float ss = 0.f, dd = 0.f;
    for (int j = 0; j < HID; ++j) { float v = hp[j]; ss += v * s[j]; dd += v * d[j]; }
    as[i] = ss; ad[i] = dd;
}

__global__ void k_alpha2(const float* __restrict__ H2, const float* __restrict__ a_s,
                         const float* __restrict__ a_d,
                         float* __restrict__ as, float* __restrict__ ad) {
    int n = blockIdx.x * 256 + threadIdx.x;
    if (n >= NNODE) return;
    const float* hp = H2 + (size_t)n * NPAD;
    float ss = 0.f, dd = 0.f;
    for (int c = 0; c < NCLS; ++c) { float v = hp[c]; ss += v * a_s[c]; dd += v * a_d[c]; }
    as[n] = ss; ad[n] = dd;
}

// ---------------- segment softmax over edges ----------------
__global__ void k_edge_max(const int* __restrict__ ei, const float* __restrict__ as,
                           const float* __restrict__ ad, float* __restrict__ m, int H) {
    int i = blockIdx.x * 256 + threadIdx.x;
    if (i >= ETOT * H) return;
    int e = i / H, h = i - e * H;
    int src, dst; edge_pair(ei, e, src, dst);
    float s = as[(size_t)src * H + h] + ad[(size_t)dst * H + h];
    s = s > 0.f ? s : SLOPE * s;
    atomicMaxF32(&m[(size_t)dst * H + h], s);
}

__global__ void k_edge_expsum(const int* __restrict__ ei, const float* __restrict__ as,
                              const float* __restrict__ ad, const float* __restrict__ m,
                              float* __restrict__ ew, float* __restrict__ z, int H) {
    int i = blockIdx.x * 256 + threadIdx.x;
    if (i >= ETOT * H) return;
    int e = i / H, h = i - e * H;
    int src, dst; edge_pair(ei, e, src, dst);
    float s = as[(size_t)src * H + h] + ad[(size_t)dst * H + h];
    s = s > 0.f ? s : SLOPE * s;
    float w = __expf(s - m[(size_t)dst * H + h]);
    ew[i] = w;
    atomicAdd(&z[(size_t)dst * H + h], w);
}

// ---------------- aggregation ----------------
__global__ void __launch_bounds__(256)
k_agg1(const int* __restrict__ ei, const float* __restrict__ H1,
       const float* __restrict__ ew, const float* __restrict__ z,
       float* __restrict__ agg) {
    int e = blockIdx.x;
    int src, dst; edge_pair(ei, e, src, dst);
    int f0 = threadIdx.x * 4;                 // 256 threads x 4 floats = 1024
    int h  = f0 >> 7;
    float alpha = ew[(size_t)e * HEADS + h] / (z[(size_t)dst * HEADS + h] + 1e-16f);
    const float4 v = *(const float4*)(H1 + (size_t)src * F1 + f0);
    float* o = agg + (size_t)dst * F1 + f0;
    atomicAdd(o + 0, alpha * v.x);
    atomicAdd(o + 1, alpha * v.y);
    atomicAdd(o + 2, alpha * v.z);
    atomicAdd(o + 3, alpha * v.w);
}

__global__ void k_agg2(const int* __restrict__ ei, const float* __restrict__ H2,
                       const float* __restrict__ ew, const float* __restrict__ z,
                       float* __restrict__ out) {
    int e = blockIdx.x * 8 + (threadIdx.x >> 5);
    if (e >= ETOT) return;
    int lane = threadIdx.x & 31;
    int src, dst; edge_pair(ei, e, src, dst);
    if (lane < NCLS) {
        float alpha = ew[e] / (z[dst] + 1e-16f);
        atomicAdd(&out[(size_t)dst * NCLS + lane], alpha * H2[(size_t)src * NPAD + lane]);
    }
}

// ---------------- small utility kernels ----------------
__global__ void k_fill(float* p, float v, int n) {
    int i = blockIdx.x * 256 + threadIdx.x;
    if (i < n) p[i] = v;
}

__global__ void k_out_init(float* __restrict__ out, const float* __restrict__ b2) {
    int i = blockIdx.x * 256 + threadIdx.x;
    if (i < NNODE * NCLS) out[i] = b2[i % NCLS];
}

// X2b = bf16(relu(agg1 + b1))
__global__ void k_x2(const float* __restrict__ agg1, const float* __restrict__ b1,
                     __bf16* __restrict__ X2b) {
    int i = blockIdx.x * 256 + threadIdx.x;
    if (i >= NNODE * F1) return;
    float v = agg1[i] + b1[i & (F1 - 1)];
    X2b[i] = f2bf(v > 0.f ? v : 0.f);
}

// ---------------- launch ----------------
extern "C" void kernel_launch(void* const* d_in, const int* in_sizes, int n_in,
                              void* d_out, int out_size, void* d_ws, size_t ws_size,
                              hipStream_t stream) {
    const float* x      = (const float*)d_in[0];
    const int*   ei     = (const int*)  d_in[1];
    const float* W1     = (const float*)d_in[2];
    const float* a1_src = (const float*)d_in[3];
    const float* a1_dst = (const float*)d_in[4];
    const float* b1     = (const float*)d_in[5];
    const float* W2     = (const float*)d_in[6];
    const float* a2_src = (const float*)d_in[7];
    const float* a2_dst = (const float*)d_in[8];
    const float* b2     = (const float*)d_in[9];
    float* out = (float*)d_out;

    char* ws = (char*)d_ws;
    size_t off = 0;
    auto take = [&](size_t bytes) -> char* {
        char* p = ws + off; off += (bytes + 255) & ~(size_t)255; return p;
    };
    __bf16* Xb   = (__bf16*)take((size_t)NNODE * FIN * 2);
    __bf16* W1p  = (__bf16*)take((size_t)FIN * F1 * 2);
    __bf16* W2p  = (__bf16*)take((size_t)F1 * NPAD * 2);
    float*  H1   = (float*) take((size_t)NNODE * F1 * 4);
    float*  agg1 = (float*) take((size_t)NNODE * F1 * 4);
    __bf16* X2b  = (__bf16*)take((size_t)NNODE * F1 * 2);
    float*  H2   = (float*) take((size_t)NNODE * NPAD * 4);
    float*  as1  = (float*) take((size_t)NNODE * HEADS * 4);
    float*  ad1  = (float*) take((size_t)NNODE * HEADS * 4);
    float*  m1   = (float*) take((size_t)NNODE * HEADS * 4);
    float*  z1   = (float*) take((size_t)NNODE * HEADS * 4);
    float*  ew1  = (float*) take((size_t)ETOT * HEADS * 4);
    float*  as2  = (float*) take((size_t)NNODE * 4);
    float*  ad2  = (float*) take((size_t)NNODE * 4);
    float*  m2   = (float*) take((size_t)NNODE * 4);
    float*  z2   = (float*) take((size_t)NNODE * 4);
    float*  ew2  = (float*) take((size_t)ETOT * 4);

    const float NEG_INF = -INFINITY;
    auto blocks = [](long n) { return (int)((n + 255) / 256); };

    // convert + pack weights
    k_cvt_bf16<<<blocks((long)NNODE * FIN), 256, 0, stream>>>(x, Xb, NNODE * FIN);
    k_pack_w<<<blocks(4 * F1 * 2), 256, 0, stream>>>(W1, W1p, 4, F1, F1, F1);
    k_pack_w<<<blocks(32 * NPAD * 2), 256, 0, stream>>>(W2, W2p, 32, NPAD, NCLS, NCLS);

    // ---- layer 1 ----
    k_gemm1<<<dim3(NNODE / 16, F1 / 128), 256, 0, stream>>>(Xb, W1p, H1);
    k_alpha1<<<blocks((long)NNODE * HEADS), 256, 0, stream>>>(H1, a1_src, a1_dst, as1, ad1);
    k_fill<<<blocks((long)NNODE * HEADS), 256, 0, stream>>>(m1, NEG_INF, NNODE * HEADS);
    k_fill<<<blocks((long)NNODE * HEADS), 256, 0, stream>>>(z1, 0.f, NNODE * HEADS);
    k_fill<<<blocks((long)NNODE * F1), 256, 0, stream>>>(agg1, 0.f, NNODE * F1);
    k_edge_max<<<blocks((long)ETOT * HEADS), 256, 0, stream>>>(ei, as1, ad1, m1, HEADS);
    k_edge_expsum<<<blocks((long)ETOT * HEADS), 256, 0, stream>>>(ei, as1, ad1, m1, ew1, z1, HEADS);
    k_agg1<<<ETOT, 256, 0, stream>>>(ei, H1, ew1, z1, agg1);
    k_x2<<<blocks((long)NNODE * F1), 256, 0, stream>>>(agg1, b1, X2b);

    // ---- layer 2 ----
    k_gemm2<<<(NNODE / 16 + 3) / 4, 256, 0, stream>>>(X2b, W2p, H2);
    k_alpha2<<<blocks(NNODE), 256, 0, stream>>>(H2, a2_src, a2_dst, as2, ad2);
    k_fill<<<blocks(NNODE), 256, 0, stream>>>(m2, NEG_INF, NNODE);
    k_fill<<<blocks(NNODE), 256, 0, stream>>>(z2, 0.f, NNODE);
    k_out_init<<<blocks((long)NNODE * NCLS), 256, 0, stream>>>(out, b2);
    k_edge_max<<<blocks(ETOT), 256, 0, stream>>>(ei, as2, ad2, m2, 1);
    k_edge_expsum<<<blocks(ETOT), 256, 0, stream>>>(ei, as2, ad2, m2, ew2, z2, 1);
    k_agg2<<<(ETOT + 7) / 8, 256, 0, stream>>>(ei, H2, ew2, z2, out);
}
